// DeltaNetBase_3367254360748
// MI455X (gfx1250) — matrix operand order repaired and checked by reading, not compiled, not run on hardware
//
#include <hip/hip_runtime.h>
#include <hip/hip_bf16.h>
#include <stdint.h>

// ---------------- problem constants (from reference) ----------------
#define NB    8
#define NPC   2048
#define NPT   (NB * NPC)      // 16384 points
#define KNBR  20
#define KNRM  10
#define REGL  0.001f
#define EPSF  1e-8f

// ---------------- CDNA5 WMMA types ----------------
typedef __bf16 bf16_t;
typedef bf16_t v16bf __attribute__((ext_vector_type(16)));
typedef bf16_t v8bf  __attribute__((ext_vector_type(8)));
typedef float  v8f   __attribute__((ext_vector_type(8)));
typedef unsigned short su8v __attribute__((ext_vector_type(8)));

__device__ __forceinline__ unsigned short f32_to_bf16_bits(float f) {
  union { float f; unsigned int u; } cv; cv.f = f;
  unsigned int u = cv.u;
  u += 0x7FFFu + ((u >> 16) & 1u);   // round-to-nearest-even
  return (unsigned short)(u >> 16);
}

// Build a 16-element bf16 fragment from two contiguous 8-element runs.
// (16-bit A layout: per lane the 16 values are two contiguous K-runs of 8;
//  pre-packed B layout: one contiguous run of 16.)
__device__ __forceinline__ v16bf load_frag2(const unsigned short* p0,
                                            const unsigned short* p1) {
  v8bf lo = __builtin_bit_cast(v8bf, *(const su8v*)p0);
  v8bf hi = __builtin_bit_cast(v8bf, *(const su8v*)p1);
  return __builtin_shufflevector(lo, hi, 0, 1, 2, 3, 4, 5, 6, 7,
                                 8, 9, 10, 11, 12, 13, 14, 15);
}

// =====================================================================
// 1) kNN: per query point, scan its batch (2048 candidates) through LDS
//    tiles, keep 20 smallest d2 (sorted ascending). top-10 = prefix.
// =====================================================================
__global__ void knn_kernel(const float* __restrict__ pos,
                           int* __restrict__ nbr, int* __restrict__ nbr_n) {
  int q = blockIdx.x * blockDim.x + threadIdx.x;   // grid covers NPT exactly
  int b = q / NPC;
  float px = pos[q * 3 + 0], py = pos[q * 3 + 1], pz = pos[q * 3 + 2];
  float bd[KNBR]; int bi[KNBR];
  for (int i = 0; i < KNBR; ++i) { bd[i] = 3.4e38f; bi[i] = q; }
  __shared__ float sp[256 * 3];
  int base = b * NPC;
  for (int t = 0; t < NPC; t += 256) {
    int c = base + t + threadIdx.x;
    sp[threadIdx.x * 3 + 0] = pos[c * 3 + 0];
    sp[threadIdx.x * 3 + 1] = pos[c * 3 + 1];
    sp[threadIdx.x * 3 + 2] = pos[c * 3 + 2];
    __syncthreads();
    for (int j = 0; j < 256; ++j) {
      float dx = px - sp[j * 3 + 0], dy = py - sp[j * 3 + 1], dz = pz - sp[j * 3 + 2];
      float d2 = dx * dx + dy * dy + dz * dz;
      if (d2 < bd[KNBR - 1]) {
        int p = KNBR - 1;
        while (p > 0 && bd[p - 1] > d2) { bd[p] = bd[p - 1]; bi[p] = bi[p - 1]; --p; }
        bd[p] = d2; bi[p] = base + t + j;
      }
    }
    __syncthreads();
  }
  for (int i = 0; i < KNBR; ++i) nbr[q * KNBR + i] = bi[i];
  for (int i = 0; i < KNRM; ++i) nbr_n[q * KNRM + i] = bi[i];
}

// =====================================================================
// 2) normal / tangent basis: 3x3 covariance, analytic smallest eigvec
// =====================================================================
__global__ void basis_kernel(const float* __restrict__ pos, const int* __restrict__ nbr_n,
                             float* __restrict__ xb, float* __restrict__ yb) {
  int n = blockIdx.x * blockDim.x + threadIdx.x;
  if (n >= NPT) return;
  float px = pos[n * 3 + 0], py = pos[n * 3 + 1], pz = pos[n * 3 + 2];
  float c00 = 0, c01 = 0, c02 = 0, c11 = 0, c12 = 0, c22 = 0;
  for (int k = 0; k < KNRM; ++k) {
    int j = nbr_n[n * KNRM + k];
    float dx = pos[j * 3 + 0] - px, dy = pos[j * 3 + 1] - py, dz = pos[j * 3 + 2] - pz;
    c00 += dx * dx; c01 += dx * dy; c02 += dx * dz;
    c11 += dy * dy; c12 += dy * dz; c22 += dz * dz;
  }
  float q = (c00 + c11 + c22) * (1.f / 3.f);
  float p1 = c01 * c01 + c02 * c02 + c12 * c12;
  float a00 = c00 - q, a11 = c11 - q, a22 = c22 - q;
  float p2 = a00 * a00 + a11 * a11 + a22 * a22 + 2.f * p1;
  float p = sqrtf(p2 * (1.f / 6.f) + 1e-30f);
  float ip = 1.f / p;
  float b00 = a00 * ip, b01 = c01 * ip, b02 = c02 * ip;
  float b11 = a11 * ip, b12 = c12 * ip, b22 = a22 * ip;
  float detB = b00 * (b11 * b22 - b12 * b12) - b01 * (b01 * b22 - b12 * b02)
             + b02 * (b01 * b12 - b11 * b02);
  float r = fminf(1.f, fmaxf(-1.f, 0.5f * detB));
  float phi = acosf(r) * (1.f / 3.f);
  float e3 = q + 2.f * p * cosf(phi + 2.0943951023931953f);  // smallest
  float m00 = c00 - e3, m11 = c11 - e3, m22 = c22 - e3;
  float v0x = c01 * c12 - c02 * m11, v0y = c02 * c01 - m00 * c12, v0z = m00 * m11 - c01 * c01;
  float v1x = c01 * m22 - c02 * c12, v1y = c02 * c02 - m00 * m22, v1z = m00 * c12 - c01 * c02;
  float v2x = m11 * m22 - c12 * c12, v2y = c12 * c02 - c01 * m22, v2z = c01 * c12 - m11 * c02;
  float s0 = v0x * v0x + v0y * v0y + v0z * v0z;
  float s1 = v1x * v1x + v1y * v1y + v1z * v1z;
  float s2 = v2x * v2x + v2y * v2y + v2z * v2z;
  float nx = v2x, ny = v2y, nz = v2z, sb = s2;
  if (s1 > sb) { nx = v1x; ny = v1y; nz = v1z; sb = s1; }
  if (s0 > sb) { nx = v0x; ny = v0y; nz = v0z; sb = s0; }
  if (sb < 1e-30f) { nx = 0.f; ny = 0.f; nz = 1.f; sb = 1.f; }
  float inv = rsqrtf(sb);
  nx *= inv; ny *= inv; nz *= inv;
  if (nx * px + ny * py + nz * pz < 0.f) { nx = -nx; ny = -ny; nz = -nz; }
  float tx, ty, tz;
  if (fabsf(nx) > 0.9f) { tx = 0.f; ty = 1.f; tz = 0.f; } else { tx = 1.f; ty = 0.f; tz = 0.f; }
  float xx = ty * nz - tz * ny, xy = tz * nx - tx * nz, xz = tx * ny - ty * nx;
  float xn = sqrtf(xx * xx + xy * xy + xz * xz) + EPSF;
  xx /= xn; xy /= xn; xz /= xn;
  float yx = ny * xz - nz * xy, yy = nz * xx - nx * xz, yz = nx * xy - ny * xx;
  xb[n * 3 + 0] = xx; xb[n * 3 + 1] = xy; xb[n * 3 + 2] = xz;
  yb[n * 3 + 0] = yx; yb[n * 3 + 1] = yy; yb[n * 3 + 2] = yz;
}

// =====================================================================
// 3) global mean(dist2) reduction, then Gw (2x20) via 3x3 adjugate solve
// =====================================================================
__global__ void h2_kernel(const float* __restrict__ pos, const int* __restrict__ nbr,
                          const float* __restrict__ xb, const float* __restrict__ yb,
                          float* __restrict__ h2sum) {
  int n = blockIdx.x * blockDim.x + threadIdx.x;
  float acc = 0.f;
  if (n < NPT) {
    float px = pos[n * 3], py = pos[n * 3 + 1], pz = pos[n * 3 + 2];
    float xx = xb[n * 3], xy = xb[n * 3 + 1], xz = xb[n * 3 + 2];
    float yx = yb[n * 3], yy = yb[n * 3 + 1], yz = yb[n * 3 + 2];
    for (int k = 0; k < KNBR; ++k) {
      int j = nbr[n * KNBR + k];
      float dx = pos[j * 3] - px, dy = pos[j * 3 + 1] - py, dz = pos[j * 3 + 2] - pz;
      float u = dx * xx + dy * xy + dz * xz;
      float v = dx * yx + dy * yy + dz * yz;
      acc += u * u + v * v;
    }
  }
  __shared__ float red[256];
  red[threadIdx.x] = acc;
  __syncthreads();
  for (int s = 128; s > 0; s >>= 1) {
    if (threadIdx.x < s) red[threadIdx.x] += red[threadIdx.x + s];
    __syncthreads();
  }
  if (threadIdx.x == 0) atomicAdd(h2sum, red[0]);
}

__global__ void grad_kernel(const float* __restrict__ pos, const int* __restrict__ nbr,
                            const float* __restrict__ xb, const float* __restrict__ yb,
                            const float* __restrict__ h2sum, float* __restrict__ Gw) {
  int n = blockIdx.x * blockDim.x + threadIdx.x;
  if (n >= NPT) return;
  float h2 = h2sum[0] / (float)(NPT * KNBR) + EPSF;  // KW=1
  float px = pos[n * 3], py = pos[n * 3 + 1], pz = pos[n * 3 + 2];
  float xx = xb[n * 3], xy = xb[n * 3 + 1], xz = xb[n * 3 + 2];
  float yx = yb[n * 3], yy = yb[n * 3 + 1], yz = yb[n * 3 + 2];
  float u[KNBR], vv[KNBR], w[KNBR];
  float a00 = REGL, a01 = 0, a02 = 0, a11 = REGL, a12 = 0, a22 = REGL;
  for (int k = 0; k < KNBR; ++k) {
    int j = nbr[n * KNBR + k];
    float dx = pos[j * 3] - px, dy = pos[j * 3 + 1] - py, dz = pos[j * 3 + 2] - pz;
    float uu = dx * xx + dy * xy + dz * xz;
    float vq = dx * yx + dy * yy + dz * yz;
    float wk = __expf(-(uu * uu + vq * vq) / h2);
    u[k] = uu; vv[k] = vq; w[k] = wk;
    a00 += wk; a01 += wk * uu; a02 += wk * vq;
    a11 += wk * uu * uu; a12 += wk * uu * vq; a22 += wk * vq * vq;
  }
  float i00 = a11 * a22 - a12 * a12;
  float i01 = a02 * a12 - a01 * a22;
  float i02 = a01 * a12 - a02 * a11;
  float i11 = a00 * a22 - a02 * a02;
  float i12 = a01 * a02 - a00 * a12;
  float i22 = a00 * a11 - a01 * a01;
  float id = 1.f / (a00 * i00 + a01 * i01 + a02 * i02);
  for (int k = 0; k < KNBR; ++k) {
    float w0 = w[k], w1 = w[k] * u[k], w2 = w[k] * vv[k];
    Gw[(n * 2 + 0) * KNBR + k] = id * (i01 * w0 + i11 * w1 + i12 * w2);
    Gw[(n * 2 + 1) * KNBR + k] = id * (i02 * w0 + i12 * w1 + i22 * w2);
  }
}

// =====================================================================
// 4) small scalar helpers
// =====================================================================
__global__ void grad_apply_kernel(const float* __restrict__ Gw, const int* __restrict__ nbr,
                                  const float* __restrict__ src, int F, float* __restrict__ dst) {
  int idx = blockIdx.x * blockDim.x + threadIdx.x;
  int total = NPT * 2 * F;
  if (idx >= total) return;
  int f = idx % F; int c = (idx / F) & 1; int n = idx / (2 * F);
  float s = 0.f;
  for (int k = 0; k < KNBR; ++k)
    s += Gw[(n * 2 + c) * KNBR + k] * src[(size_t)nbr[n * KNBR + k] * F + f];
  dst[(size_t)(n * 2 + c) * F + f] = s;
}

__global__ void center_kernel(const float* __restrict__ pos, const int* __restrict__ nbr,
                              float* __restrict__ xc) {
  int n = blockIdx.x * blockDim.x + threadIdx.x;
  if (n >= NPT) return;
  float m0 = 0, m1 = 0, m2 = 0;
  for (int k = 0; k < KNBR; ++k) {
    int j = nbr[n * KNBR + k];
    m0 += pos[j * 3]; m1 += pos[j * 3 + 1]; m2 += pos[j * 3 + 2];
  }
  const float s = 1.f / KNBR;
  xc[n * 3 + 0] = pos[n * 3 + 0] - m0 * s;
  xc[n * 3 + 1] = pos[n * 3 + 1] - m1 * s;
  xc[n * 3 + 2] = pos[n * 3 + 2] - m2 * s;
}

__global__ void div_curl_kernel(const float* __restrict__ Gw, const int* __restrict__ nbr,
                                const float* __restrict__ v, int F,
                                float* __restrict__ divv, float* __restrict__ curl) {
  int idx = blockIdx.x * blockDim.x + threadIdx.x;
  int total = NPT * KNBR * F;
  if (idx >= total) return;
  int f = idx % F; int k = (idx / F) % KNBR; int n = idx / (F * KNBR);
  float g0 = Gw[(n * 2 + 0) * KNBR + k], g1 = Gw[(n * 2 + 1) * KNBR + k];
  float v0 = v[(size_t)(n * 2 + 0) * F + f], v1 = v[(size_t)(n * 2 + 1) * F + f];
  int j = nbr[n * KNBR + k];
  atomicAdd(&divv[(size_t)j * F + f], -(g0 * v0 + g1 * v1));
  atomicAdd(&curl[(size_t)j * F + f], -(g0 * (-v1) + g1 * v0));  // rot(v)
}

__global__ void vnorm_kernel(float* __restrict__ vlin, const float* __restrict__ bv, int co) {
  int idx = blockIdx.x * blockDim.x + threadIdx.x;
  if (idx >= NPT * co) return;
  int o = idx % co, n = idx / co;
  float a = vlin[(size_t)(n * 2 + 0) * co + o];
  float b = vlin[(size_t)(n * 2 + 1) * co + o];
  float nrm = sqrtf(a * a + b * b + EPSF);
  float s = fmaxf(nrm + bv[o], 0.f) / nrm;
  vlin[(size_t)(n * 2 + 0) * co + o] = a * s;
  vlin[(size_t)(n * 2 + 1) * co + o] = b * s;
}

// =====================================================================
// 5) weight pre-pack: f32 (Kd x Nc) -> bf16 fragment layout so each
//    lane's B fragment is 16 contiguous shorts (32B = 2x b128 loads):
//    P[((kt*(Nc/16)+nt)*32 + lane)*16 + e] = W[(kt*32 + (e&7) + 8*hi + 16*(e>>3)) * Nc + nt*16 + rlo]
// =====================================================================
__global__ void pack_w_kernel(const float* __restrict__ W, int Nc,
                              unsigned short* __restrict__ P, int total) {
  int idx = blockIdx.x * blockDim.x + threadIdx.x;
  if (idx >= total) return;
  int e = idx & 15;
  int lane = (idx >> 4) & 31;
  int tile = idx >> 9;
  int ntiles = Nc >> 4;
  int kt = tile / ntiles, nt = tile % ntiles;
  int hi = lane >> 4, rlo = lane & 15;
  int k = kt * 32 + (e & 7) + hi * 8 + ((e >> 3) << 4);
  int col = nt * 16 + rlo;
  P[idx] = f32_to_bf16_bits(W[(size_t)k * Nc + col]);
}

// =====================================================================
// 6) layer-0 scalar path (ci = 3; K-dim too small for WMMA)
// =====================================================================
__global__ void edge0_kernel(const float* __restrict__ xc, const int* __restrict__ nbr,
                             const float* __restrict__ Ws, const float* __restrict__ bs,
                             float* __restrict__ xs) {
  int idx = blockIdx.x * blockDim.x + threadIdx.x;
  if (idx >= NPT * 64) return;
  int o = idx % 64, n = idx / 64;
  float xi0 = xc[n * 3], xi1 = xc[n * 3 + 1], xi2 = xc[n * 3 + 2];
  float w0 = Ws[0 * 64 + o], w1 = Ws[1 * 64 + o], w2 = Ws[2 * 64 + o];
  float w3 = Ws[3 * 64 + o], w4 = Ws[4 * 64 + o], w5 = Ws[5 * 64 + o];
  float fixed = bs[o] + xi0 * w0 + xi1 * w1 + xi2 * w2;
  float best = 0.f;   // relu outputs are >= 0
  for (int k = 0; k < KNBR; ++k) {
    int j = nbr[n * KNBR + k];
    float h = fixed + (xc[j * 3] - xi0) * w3 + (xc[j * 3 + 1] - xi1) * w4 + (xc[j * 3 + 2] - xi2) * w5;
    best = fmaxf(best, fmaxf(h, 0.f));
  }
  xs[n * 64 + o] = best;
}

__global__ void wc0_kernel(const float* __restrict__ xs, const float* __restrict__ divv,
                           const float* __restrict__ curl, const float* __restrict__ Wc,
                           const float* __restrict__ bc, float* __restrict__ y) {
  int idx = blockIdx.x * blockDim.x + threadIdx.x;
  if (idx >= NPT * 64) return;
  int o = idx % 64, n = idx / 64;
  float s = bc[o];
  for (int f = 0; f < 64; ++f) s += xs[n * 64 + f] * Wc[f * 64 + o];
  for (int f = 0; f < 3; ++f)  s += divv[n * 3 + f] * Wc[(64 + f) * 64 + o];
  for (int f = 0; f < 3; ++f)  s += curl[n * 3 + f] * Wc[(67 + f) * 64 + o];
  y[n * 64 + o] = fmaxf(s, 0.f);
}

__global__ void wv0_kernel(const float* __restrict__ v, const float* __restrict__ gy,
                           const float* __restrict__ Wv, float* __restrict__ vlin) {
  int idx = blockIdx.x * blockDim.x + threadIdx.x;
  if (idx >= NPT * 2 * 64) return;
  int o = idx % 64; int c = (idx / 64) & 1; int n = idx / 128;
  float s = 0.f;
  for (int f = 0; f < 3; ++f) {
    float vf = v[(n * 2 + c) * 3 + f];
    float rv = (c == 0) ? -v[(n * 2 + 1) * 3 + f] : v[(n * 2 + 0) * 3 + f];
    s += vf * Wv[f * 64 + o] + rv * Wv[(3 + f) * 64 + o];
  }
  for (int f = 0; f < 64; ++f) {
    float g  = gy[(n * 2 + c) * 64 + f];
    float rg = (c == 0) ? -gy[(n * 2 + 1) * 64 + f] : gy[(n * 2 + 0) * 64 + f];
    s += g * Wv[(6 + f) * 64 + o] + rg * Wv[(70 + f) * 64 + o];
  }
  vlin[(n * 2 + c) * 64 + o] = s;
}

// =====================================================================
// 7) WMMA edge MLP (layers 1..3): one block per point, compile-time
//    shapes, fully unrolled WMMA chain, vectorized fragment loads.
// =====================================================================
template <int CI, int CO>
__global__ void edge_wmma_kernel(const float* __restrict__ x, const int* __restrict__ nbr,
                                 const unsigned short* __restrict__ Wp,
                                 const float* __restrict__ bs,
                                 float* __restrict__ xs) {
  constexpr int KD = 2 * CI;
  constexpr int NT = CO / 16;       // n tiles
  constexpr int KT = KD / 32;       // k steps
  __shared__ unsigned short Esh[32 * KD];
  __shared__ float Hsh[32 * CO];
  int n = blockIdx.x;
  int tid = threadIdx.x;
  // stage E = [xi | xj - xi] (rows 20..31 padded with row n, excluded from max)
  for (int idx = tid; idx < 32 * KD; idx += 256) {
    int r = idx / KD, c = idx % KD;
    float val;
    if (c < CI) {
      val = x[(size_t)n * CI + c];
    } else {
      int j = (r < KNBR) ? nbr[n * KNBR + r] : n;
      int f = c - CI;
      val = x[(size_t)j * CI + f] - x[(size_t)n * CI + f];
    }
    Esh[r * KD + c] = f32_to_bf16_bits(val);
  }
  __syncthreads();
  int wave = tid >> 5, lane = tid & 31;
  int rlo = lane & 15;
  int hi = lane >> 4;
  int hiOff = hi * 8;
  for (int t = wave; t < 2 * NT; t += 8) {
    int mt = t / NT, nt = t % NT;
    const unsigned short* arow = &Esh[(mt * 16 + rlo) * KD + hi * 8];
    const unsigned short* bcol = Wp + ((size_t)nt * 32 + lane) * 16;
    const size_t bstride = (size_t)NT * 32 * 16;
    v8f acc{};
#pragma unroll
    for (int kt = 0; kt < KT; ++kt) {
      v16bf a = load_frag2(arow + kt * 32, arow + kt * 32 + 16);
      const unsigned short* bp = bcol + kt * bstride;
      v16bf b = load_frag2(bp, bp + 8);
      acc = __builtin_amdgcn_wmma_f32_16x16x32_bf16(false, a, false, b, (short)0, acc,
                                                    false, false);
    }
    float bb = bs[nt * 16 + rlo];
#pragma unroll
    for (int r = 0; r < 8; ++r) {
      int mrow = mt * 16 + r + hiOff;               // C: M = r + 8*hi
      Hsh[mrow * CO + nt * 16 + rlo] = fmaxf(acc[r] + bb, 0.f);
    }
  }
  __syncthreads();
  for (int c = tid; c < CO; c += 256) {
    float m = Hsh[c];
#pragma unroll
    for (int r = 1; r < KNBR; ++r) m = fmaxf(m, Hsh[r * CO + c]);
    xs[(size_t)n * CO + c] = m;
  }
}

// =====================================================================
// 8) generic bf16 WMMA GEMM: C[M x Nc] = act(A[M x KD] @ W + b)
//    A row-major bf16, W pre-packed; one wave per 16x16 tile; K unrolled.
// =====================================================================
template <int KD>
__global__ void gemm_wmma_kernel(const unsigned short* __restrict__ A,
                                 const unsigned short* __restrict__ Wp,
                                 const float* __restrict__ bias,
                                 float* __restrict__ C,
                                 int Nc, int relu) {
  int mt = blockIdx.x, nt = blockIdx.y;
  int lane = threadIdx.x;
  int rlo = lane & 15;
  int hi = lane >> 4;
  int hiOff = hi * 8;
  int ntiles = Nc >> 4;
  const unsigned short* arow = A + (size_t)(mt * 16 + rlo) * KD + hi * 8;
  const unsigned short* bcol = Wp + ((size_t)nt * 32 + lane) * 16;
  const size_t bstride = (size_t)ntiles * 32 * 16;
  v8f acc{};
#pragma unroll
  for (int kt = 0; kt < KD / 32; ++kt) {
    v16bf a = load_frag2(arow + kt * 32, arow + kt * 32 + 16);
    const unsigned short* bp = bcol + kt * bstride;
    v16bf b = load_frag2(bp, bp + 8);
    acc = __builtin_amdgcn_wmma_f32_16x16x32_bf16(false, a, false, b, (short)0, acc,
                                                  false, false);
  }
  float bb = bias ? bias[nt * 16 + rlo] : 0.f;
#pragma unroll
  for (int r = 0; r < 8; ++r) {
    int mrow = mt * 16 + r + hiOff;
    float v = acc[r] + bb;
    if (relu) v = fmaxf(v, 0.f);
    C[(size_t)mrow * Nc + nt * 16 + rlo] = v;
  }
}

// =====================================================================
// 9) pack kernels: build bf16 row-major GEMM A inputs
// =====================================================================
__global__ void pack_cat3_kernel(const float* __restrict__ xs, const float* __restrict__ divv,
                                 const float* __restrict__ curl, int ci, int co,
                                 unsigned short* __restrict__ A) {
  int Ct = co + 2 * ci;
  int idx = blockIdx.x * blockDim.x + threadIdx.x;
  if (idx >= NPT * Ct) return;
  int c = idx % Ct, n = idx / Ct;
  float v;
  if (c < co)            v = xs[(size_t)n * co + c];
  else if (c < co + ci)  v = divv[(size_t)n * ci + (c - co)];
  else                   v = curl[(size_t)n * ci + (c - co - ci)];
  A[(size_t)n * Ct + c] = f32_to_bf16_bits(v);
}

// vf = [v | rot(v) | gy | rot(gy)], rows = 2N (n-major, c inner). gy computed inline.
__global__ void pack_vf_kernel(const float* __restrict__ v, const float* __restrict__ y,
                               const float* __restrict__ Gw, const int* __restrict__ nbr,
                               int ci, int co, unsigned short* __restrict__ A) {
  int Ct = 2 * ci + 2 * co;
  int idx = blockIdx.x * blockDim.x + threadIdx.x;
  if (idx >= NPT * 2 * Ct) return;
  int c = idx % Ct; int row = idx / Ct; int cx = row & 1; int n = row >> 1;
  float val;
  if (c < ci) {
    val = v[(size_t)(n * 2 + cx) * ci + c];
  } else if (c < 2 * ci) {
    int f = c - ci;
    val = (cx == 0) ? -v[(size_t)(n * 2 + 1) * ci + f] : v[(size_t)(n * 2 + 0) * ci + f];
  } else {
    int f, gc; float sgn;
    if (c < 2 * ci + co) { f = c - 2 * ci; gc = cx; sgn = 1.f; }
    else { f = c - 2 * ci - co; gc = 1 - cx; sgn = (cx == 0) ? -1.f : 1.f; }
    float s = 0.f;
    for (int k = 0; k < KNBR; ++k)
      s += Gw[(n * 2 + gc) * KNBR + k] * y[(size_t)nbr[n * KNBR + k] * co + f];
    val = sgn * s;
  }
  A[(size_t)row * Ct + c] = f32_to_bf16_bits(val);
}

// =====================================================================
// host orchestration
// =====================================================================
static inline int cdiv(int a, int b) { return (a + b - 1) / b; }

extern "C" void kernel_launch(void* const* d_in, const int* in_sizes, int n_in,
                              void* d_out, int out_size, void* d_ws, size_t ws_size,
                              hipStream_t stream) {
  (void)in_sizes; (void)n_in; (void)out_size; (void)ws_size;
  const float* pos = (const float*)d_in[0];
  const float* Ws[4] = {(const float*)d_in[2],  (const float*)d_in[8],
                        (const float*)d_in[14], (const float*)d_in[20]};
  const float* bs[4] = {(const float*)d_in[3],  (const float*)d_in[9],
                        (const float*)d_in[15], (const float*)d_in[21]};
  const float* Wc[4] = {(const float*)d_in[4],  (const float*)d_in[10],
                        (const float*)d_in[16], (const float*)d_in[22]};
  const float* bc[4] = {(const float*)d_in[5],  (const float*)d_in[11],
                        (const float*)d_in[17], (const float*)d_in[23]};
  const float* Wv[3] = {(const float*)d_in[6],  (const float*)d_in[12],
                        (const float*)d_in[18]};
  const float* bv[3] = {(const float*)d_in[7],  (const float*)d_in[13],
                        (const float*)d_in[19]};
  const int ci_[4] = {3, 64, 64, 128};
  const int co_[4] = {64, 64, 128, 256};

  // ---- scratch carve ----
  char* wsp = (char*)d_ws;
  size_t off = 0;
  auto carve = [&](size_t bytes) -> void* {
    void* p = wsp + off;
    off = (off + bytes + 255) & ~(size_t)255;
    return p;
  };
  int*   nbr   = (int*)  carve((size_t)NPT * KNBR * 4);
  int*   nbr_n = (int*)  carve((size_t)NPT * KNRM * 4);
  float* xb    = (float*)carve((size_t)NPT * 3 * 4);
  float* yb    = (float*)carve((size_t)NPT * 3 * 4);
  float* Gw    = (float*)carve((size_t)NPT * 2 * KNBR * 4);
  float* h2s   = (float*)carve(256);
  float* xc    = (float*)carve((size_t)NPT * 3 * 4);
  float* vA    = (float*)carve((size_t)NPT * 2 * 128 * 4);
  float* vB    = (float*)carve((size_t)NPT * 2 * 128 * 4);
  float* xsb   = (float*)carve((size_t)NPT * 256 * 4);
  float* divv  = (float*)carve((size_t)NPT * 128 * 4);   // reused as gy in layer 0
  float* curl  = (float*)carve((size_t)NPT * 128 * 4);
  unsigned short* bigbf = (unsigned short*)carve((size_t)2 * NPT * 384 * 2);
  unsigned short* Wsb[4] = {nullptr, nullptr, nullptr, nullptr};
  unsigned short* Wcb[4] = {nullptr, nullptr, nullptr, nullptr};
  unsigned short* Wvb[3] = {nullptr, nullptr, nullptr};
  for (int i = 1; i < 4; ++i) {
    Wsb[i] = (unsigned short*)carve((size_t)2 * ci_[i] * co_[i] * 2);
    Wcb[i] = (unsigned short*)carve((size_t)(co_[i] + 2 * ci_[i]) * co_[i] * 2);
  }
  for (int i = 1; i < 3; ++i)
    Wvb[i] = (unsigned short*)carve((size_t)(2 * ci_[i] + 2 * co_[i]) * co_[i] * 2);

  float* out = (float*)d_out;
  float* y_out[4] = {out, out + (size_t)NPT * 64, out + (size_t)NPT * 128,
                     out + (size_t)NPT * 256};

  // ---- weight conversion into fragment-packed bf16 ----
  for (int i = 1; i < 4; ++i) {
    int n1 = 2 * ci_[i] * co_[i];
    pack_w_kernel<<<cdiv(n1, 256), 256, 0, stream>>>(Ws[i], co_[i], Wsb[i], n1);
    int n2 = (co_[i] + 2 * ci_[i]) * co_[i];
    pack_w_kernel<<<cdiv(n2, 256), 256, 0, stream>>>(Wc[i], co_[i], Wcb[i], n2);
  }
  for (int i = 1; i < 3; ++i) {
    int n3 = (2 * ci_[i] + 2 * co_[i]) * co_[i];
    pack_w_kernel<<<cdiv(n3, 256), 256, 0, stream>>>(Wv[i], co_[i], Wvb[i], n3);
  }

  // ---- geometry ----
  knn_kernel<<<NPT / 256, 256, 0, stream>>>(pos, nbr, nbr_n);
  basis_kernel<<<cdiv(NPT, 256), 256, 0, stream>>>(pos, nbr_n, xb, yb);
  hipMemsetAsync(h2s, 0, 256, stream);
  h2_kernel<<<cdiv(NPT, 256), 256, 0, stream>>>(pos, nbr, xb, yb, h2s);
  grad_kernel<<<cdiv(NPT, 256), 256, 0, stream>>>(pos, nbr, xb, yb, h2s, Gw);

  // initial v = grad_apply(Gw, nbr, pos), F=3
  grad_apply_kernel<<<cdiv(NPT * 2 * 3, 256), 256, 0, stream>>>(Gw, nbr, pos, 3, vA);

  // ---- layer 0 (scalar; ci=3) ----
  center_kernel<<<cdiv(NPT, 256), 256, 0, stream>>>(pos, nbr, xc);
  edge0_kernel<<<cdiv(NPT * 64, 256), 256, 0, stream>>>(xc, nbr, Ws[0], bs[0], xsb);
  hipMemsetAsync(divv, 0, (size_t)NPT * 3 * 4, stream);
  hipMemsetAsync(curl, 0, (size_t)NPT * 3 * 4, stream);
  div_curl_kernel<<<cdiv(NPT * KNBR * 3, 256), 256, 0, stream>>>(Gw, nbr, vA, 3, divv, curl);
  wc0_kernel<<<cdiv(NPT * 64, 256), 256, 0, stream>>>(xsb, divv, curl, Wc[0], bc[0], y_out[0]);
  grad_apply_kernel<<<cdiv(NPT * 2 * 64, 256), 256, 0, stream>>>(Gw, nbr, y_out[0], 64, divv);
  wv0_kernel<<<cdiv(NPT * 2 * 64, 256), 256, 0, stream>>>(vA, divv, Wv[0], vB);
  vnorm_kernel<<<cdiv(NPT * 64, 256), 256, 0, stream>>>(vB, bv[0], 64);

  // ---- layers 1..3 (WMMA path) ----
  float* vcur = vB;
  float* vnext = vA;
  for (int i = 1; i < 4; ++i) {
    int ci = ci_[i], co = co_[i];
    const float* xin = y_out[i - 1];
    if (i == 1)      edge_wmma_kernel<64, 64>  <<<NPT, 256, 0, stream>>>(xin, nbr, Wsb[i], bs[i], xsb);
    else if (i == 2) edge_wmma_kernel<64, 128> <<<NPT, 256, 0, stream>>>(xin, nbr, Wsb[i], bs[i], xsb);
    else             edge_wmma_kernel<128, 256><<<NPT, 256, 0, stream>>>(xin, nbr, Wsb[i], bs[i], xsb);

    hipMemsetAsync(divv, 0, (size_t)NPT * ci * 4, stream);
    hipMemsetAsync(curl, 0, (size_t)NPT * ci * 4, stream);
    div_curl_kernel<<<cdiv(NPT * KNBR * ci, 256), 256, 0, stream>>>(Gw, nbr, vcur, ci, divv, curl);

    int Ct = co + 2 * ci;                    // 192, 256, 512
    pack_cat3_kernel<<<cdiv(NPT * Ct, 256), 256, 0, stream>>>(xsb, divv, curl, ci, co, bigbf);
    {
      dim3 g(NPT / 16, co / 16);
      if (Ct == 192)      gemm_wmma_kernel<192><<<g, 32, 0, stream>>>(bigbf, Wcb[i], bc[i], y_out[i], co, 1);
      else if (Ct == 256) gemm_wmma_kernel<256><<<g, 32, 0, stream>>>(bigbf, Wcb[i], bc[i], y_out[i], co, 1);
      else                gemm_wmma_kernel<512><<<g, 32, 0, stream>>>(bigbf, Wcb[i], bc[i], y_out[i], co, 1);
    }
    if (i < 3) {
      int Cv = 2 * ci + 2 * co;              // 256, 384
      pack_vf_kernel<<<cdiv(NPT * 2 * Cv, 256), 256, 0, stream>>>(vcur, y_out[i], Gw, nbr,
                                                                  ci, co, bigbf);
      dim3 g((2 * NPT) / 16, co / 16);
      if (Cv == 256) gemm_wmma_kernel<256><<<g, 32, 0, stream>>>(bigbf, Wvb[i], nullptr, vnext, co, 0);
      else           gemm_wmma_kernel<384><<<g, 32, 0, stream>>>(bigbf, Wvb[i], nullptr, vnext, co, 0);
      vnorm_kernel<<<cdiv(NPT * co, 256), 256, 0, stream>>>(vnext, bv[i], co);
      float* t = vcur; vcur = vnext; vnext = t;
    }
  }
}
